// GraphNeuralNetworkGCP_23828478558384
// MI455X (gfx1250) — matrix operand order repaired and checked by reading, not compiled, not run on hardware
//
#include <hip/hip_runtime.h>
#include <hip/hip_bf16.h>
#include <math.h>

#define Bb 64
#define Nn 512
#define Cc 16
#define Dd 64
#define Tt 32
#define Hh 100
#define HP 128   // H padded to multiple of 32 for WMMA K-loops
#define G4 256   // 4*D gate width

typedef __attribute__((ext_vector_type(16))) _Float16 v16h;
typedef __attribute__((ext_vector_type(8)))  _Float16 v8h;
typedef __attribute__((ext_vector_type(8)))  float    v8f;

// ---------------- WMMA fragment helpers (wave32, 16x16x32 f16) ----------------
// A (16x32 f16, row-major src): lane l holds row l%16;
// element e -> K = (l/16)*16 + e
// Per lane that is two contiguous 8-half (16B) runs -> two b128 loads.
__device__ __forceinline__ v16h ld_a_h(const _Float16* Ap, int lda, int lane) {
  int m = lane & 15, hf = lane >> 4;
  const _Float16* base = Ap + (size_t)m * lda + hf * 16;
  v8h lo = *(const v8h*)(base);        // K = hf*16 .. hf*16+7
  v8h hi = *(const v8h*)(base + 8);    // K = hf*16+8 .. hf*16+15
  return __builtin_shufflevector(lo, hi, 0, 1, 2, 3, 4, 5, 6, 7,
                                 8, 9, 10, 11, 12, 13, 14, 15);
}
// B fragment from PRE-PACKED weights: packed[((nt*KT+kt)*32+lane)*16 + e]
// where element e -> K = kt*32 + (lane/16)*16 + e, col = nt*16 + lane%16.
// One 32-byte contiguous per-lane load.
__device__ __forceinline__ v16h ld_b_p(const _Float16* Wp, int KT, int kt, int nt, int lane) {
  return *(const v16h*)(Wp + ((((size_t)nt * KT + kt) * 32 + lane) << 4));
}
__device__ __forceinline__ float sigm(float x) { return 1.f / (1.f + expf(-x)); }

#define WMMA(a, b, c) __builtin_amdgcn_wmma_f32_16x16x32_f16(false, (a), false, (b), (short)0, (c), false, false)

// ---------------- prep kernels ----------------
__global__ void k_cvt_f16(_Float16* dst, const float* src, int n) {
  for (int i = blockIdx.x * blockDim.x + threadIdx.x; i < n; i += gridDim.x * blockDim.x)
    dst[i] = (_Float16)src[i];
}
__global__ void k_copy_f32(float* dst, const float* src, int n) {
  for (int i = blockIdx.x * blockDim.x + threadIdx.x; i < n; i += gridDim.x * blockDim.x)
    dst[i] = src[i];
}
__global__ void k_zero_f32(float* dst, int n) {
  for (int i = blockIdx.x * blockDim.x + threadIdx.x; i < n; i += gridDim.x * blockDim.x)
    dst[i] = 0.f;
}
__global__ void k_pad_f32vec(float* dst, const float* src, int sn, int dn) {
  for (int i = blockIdx.x * blockDim.x + threadIdx.x; i < dn; i += gridDim.x * blockDim.x)
    dst[i] = (i < sn) ? src[i] : 0.f;
}
__global__ void k_addvec(float* dst, const float* a, const float* b, int n) {
  for (int i = blockIdx.x * blockDim.x + threadIdx.x; i < n; i += gridDim.x * blockDim.x)
    dst[i] = a[i] + b[i];
}
// Pack f32 weight [srows,scols] into B-fragment layout for logical [K,NC] (zero padded).
__global__ void k_pack_b(_Float16* dst, const float* src, int K, int NC, int srows, int scols) {
  int KT = K / 32;
  int tot = K * NC;
  for (int i = blockIdx.x * blockDim.x + threadIdx.x; i < tot; i += gridDim.x * blockDim.x) {
    int e = i & 15;
    int lane = (i >> 4) & 31;
    int rem = i >> 9;
    int kt = rem % KT, nt = rem / KT;
    int k = kt * 32 + ((lane >> 4) << 4) + e;
    int c = nt * 16 + (lane & 15);
    dst[i] = (k < srows && c < scols) ? (_Float16)src[(size_t)k * scols + c] : (_Float16)0.f;
  }
}
// Pack f32 activation [B,N,D] into per-batch B-fragment layout (K=N, NC=D).
__global__ void k_pack_act(_Float16* dst, const float* src) {
  int KT = Nn / 32;
  int per = Nn * Dd;
  int tot = Bb * per;
  for (int i = blockIdx.x * blockDim.x + threadIdx.x; i < tot; i += gridDim.x * blockDim.x) {
    int b = i / per, j = i % per;
    int e = j & 15;
    int lane = (j >> 4) & 31;
    int rem = j >> 9;
    int kt = rem % KT, nt = rem / KT;
    int k = kt * 32 + ((lane >> 4) << 4) + e;
    int c = nt * 16 + (lane & 15);
    dst[i] = (_Float16)src[((size_t)b * Nn + k) * Dd + c];
  }
}

// ---------------- mv = Mvv @ vh1 (per batch, [512,512]@[512,64]) ----------------
// grid = B*N/16 blocks, 128 threads (4 waves); wave w computes 16x16 col tile w.
__global__ __launch_bounds__(128) void k_adj(const _Float16* __restrict__ mvvh,
                                             const _Float16* __restrict__ vh1p,
                                             _Float16* __restrict__ mvh) {
  int tile = blockIdx.x;            // global 16-row tile id
  int b = tile >> 5;
  int lane = threadIdx.x & 31, wave = threadIdx.x >> 5;
  const _Float16* Ap = mvvh + (size_t)b * Nn * Nn + (size_t)(tile & 31) * 16 * Nn;
  const _Float16* Bp = vh1p + (size_t)b * Nn * Dd;   // packed, KT = N/32
  v8f acc = {};
  for (int kt = 0; kt < Nn / 32; ++kt) {
    __builtin_prefetch(Ap + (kt + 2) * 32, 0, 3);
    v16h a = ld_a_h(Ap + kt * 32, Nn, lane);
    v16h bb = ld_b_p(Bp, Nn / 32, kt, wave, lane);
    acc = WMMA(a, bb, acc);
  }
  _Float16* Cp = mvh + (size_t)tile * 16 * Dd + wave * 16;
  int n = lane & 15, hf = lane >> 4;
#pragma unroll
  for (int e = 0; e < 8; ++e) Cp[(size_t)(hf * 8 + e) * Dd + n] = (_Float16)acc[e];
}

// ---------------- vmsg MLP + per-tile column sums ----------------
// grid = B*N/16 blocks, 128 threads. Writes partial[tile][d] = sum over 16 rows.
__global__ __launch_bounds__(128) void k_vmsg(const _Float16* __restrict__ vh1h,
                                              const _Float16* __restrict__ vW1p,
                                              const float* __restrict__ vb1p,
                                              const _Float16* __restrict__ vW2p,
                                              const float* __restrict__ vb2,
                                              float* __restrict__ partial) {
  __shared__ __align__(16) _Float16 h1s[16][HP];
  __shared__ float red[4][32];
  int tile = blockIdx.x;
  int lane = threadIdx.x & 31, wave = threadIdx.x >> 5;
  int n = lane & 15, hf = lane >> 4;
  const _Float16* Ap = vh1h + (size_t)tile * 16 * Dd;
  for (int ct = wave; ct < HP / 16; ct += 4) {
    v8f acc = {};
#pragma unroll
    for (int kt = 0; kt < Dd / 32; ++kt) {
      v16h a = ld_a_h(Ap + kt * 32, Dd, lane);
      v16h bb = ld_b_p(vW1p, Dd / 32, kt, ct, lane);
      acc = WMMA(a, bb, acc);
    }
    float bias = vb1p[ct * 16 + n];
#pragma unroll
    for (int e = 0; e < 8; ++e) {
      float v = acc[e] + bias; v = v > 0.f ? v : 0.f;
      h1s[hf * 8 + e][ct * 16 + n] = (_Float16)v;
    }
  }
  __syncthreads();
  v8f acc = {};
#pragma unroll
  for (int kt = 0; kt < HP / 32; ++kt) {
    v16h a = ld_a_h(&h1s[0][0] + kt * 32, HP, lane);
    v16h bb = ld_b_p(vW2p, HP / 32, kt, wave, lane);
    acc = WMMA(a, bb, acc);
  }
  float bias = vb2[wave * 16 + n];
  float s = 0.f;
#pragma unroll
  for (int e = 0; e < 8; ++e) { float v = acc[e] + bias; v = v > 0.f ? v : 0.f; s += v; }
  red[wave][lane] = s;
  __syncthreads();
  if (lane < 16)
    partial[(size_t)tile * Dd + wave * 16 + n] = red[wave][n] + red[wave][n + 16];
}

__global__ void k_red(const float* __restrict__ partial, float* __restrict__ vsum) {
  int b = blockIdx.x, d = threadIdx.x;
  float s = 0.f;
  for (int t = 0; t < Nn / 16; ++t) s += partial[(size_t)(b * (Nn / 16) + t) * Dd + d];
  vsum[(size_t)b * Dd + d] = s;
}

// ---------------- color pipeline: cmsg MLP, mc, vbias = mc@Wih0[:D]+b, color LSTM ----------------
// grid = B blocks, 128 threads (4 waves).
__global__ __launch_bounds__(128) void k_color(
    const float* __restrict__ ch_old, const float* __restrict__ cc_old,
    float* __restrict__ ch_new, float* __restrict__ cc_new,
    const float* __restrict__ vsum, const int* __restrict__ n_colors,
    const _Float16* __restrict__ cW1p, const float* __restrict__ cb1p,
    const _Float16* __restrict__ cW2p, const float* __restrict__ cb2,
    const _Float16* __restrict__ Wihc_p, const _Float16* __restrict__ Whhc_p,
    const float* __restrict__ bihc, const float* __restrict__ bhhc,
    const float* __restrict__ Wih0, const float* __restrict__ bih0,
    const float* __restrict__ bhh0, float* __restrict__ vbias) {
  __shared__ __align__(16) _Float16 chh[Cc][Dd];
  __shared__ __align__(16) _Float16 h1s[Cc][HP];
  __shared__ float    cmsg[Cc][Dd];
  __shared__ __align__(16) _Float16 cin[Cc][Dd];
  __shared__ float    mcs[Dd];
  __shared__ float    gates[Cc][G4];
  int b = blockIdx.x;
  int tid = threadIdx.x, lane = tid & 31, wave = tid >> 5;
  int n = lane & 15, hf = lane >> 4;
  const float* chb = ch_old + (size_t)b * Cc * Dd;
  for (int i = tid; i < Cc * Dd; i += 128) chh[0][i] = (_Float16)chb[i];
  __syncthreads();
  // cmsg layer 1
  for (int ct = wave; ct < HP / 16; ct += 4) {
    v8f acc = {};
#pragma unroll
    for (int kt = 0; kt < Dd / 32; ++kt) {
      v16h a = ld_a_h(&chh[0][0] + kt * 32, Dd, lane);
      v16h bb = ld_b_p(cW1p, Dd / 32, kt, ct, lane);
      acc = WMMA(a, bb, acc);
    }
    float bias = cb1p[ct * 16 + n];
#pragma unroll
    for (int e = 0; e < 8; ++e) {
      float v = acc[e] + bias; v = v > 0.f ? v : 0.f;
      h1s[hf * 8 + e][ct * 16 + n] = (_Float16)v;
    }
  }
  __syncthreads();
  // cmsg layer 2
  {
    v8f acc = {};
#pragma unroll
    for (int kt = 0; kt < HP / 32; ++kt) {
      v16h a = ld_a_h(&h1s[0][0] + kt * 32, HP, lane);
      v16h bb = ld_b_p(cW2p, HP / 32, kt, wave, lane);
      acc = WMMA(a, bb, acc);
    }
    float bias = cb2[wave * 16 + n];
#pragma unroll
    for (int e = 0; e < 8; ++e) {
      float v = acc[e] + bias;
      cmsg[hf * 8 + e][wave * 16 + n] = v > 0.f ? v : 0.f;
    }
  }
  __syncthreads();
  int nc = n_colors[b];
  if (tid < Dd) {
    float s = 0.f;
    for (int j = 0; j < nc; ++j) s += cmsg[j][tid];
    mcs[tid] = s;
  }
  for (int i = tid; i < Cc * Dd; i += 128) {
    int j = i / Dd, d = i % Dd;
    cin[j][d] = (_Float16)((j < nc) ? vsum[(size_t)b * Dd + d] : 0.f);
  }
  __syncthreads();
  // vertex-LSTM input bias: mc @ Wih0[:D,:] + bih0 + bhh0   (same for all vertices of batch b)
  for (int g = tid; g < G4; g += 128) {
    float acc = bih0[g] + bhh0[g];
    for (int d = 0; d < Dd; ++d) acc += mcs[d] * Wih0[(size_t)d * G4 + g];
    vbias[(size_t)b * G4 + g] = acc;
  }
  // color LSTM gates: cin@Wihc + ch@Whhc
  for (int ct = wave * 4; ct < wave * 4 + 4; ++ct) {
    v8f acc = {};
#pragma unroll
    for (int kt = 0; kt < Dd / 32; ++kt) {
      v16h a = ld_a_h(&cin[0][0] + kt * 32, Dd, lane);
      v16h bb = ld_b_p(Wihc_p, Dd / 32, kt, ct, lane);
      acc = WMMA(a, bb, acc);
      a = ld_a_h(&chh[0][0] + kt * 32, Dd, lane);
      bb = ld_b_p(Whhc_p, Dd / 32, kt, ct, lane);
      acc = WMMA(a, bb, acc);
    }
#pragma unroll
    for (int e = 0; e < 8; ++e) gates[hf * 8 + e][ct * 16 + n] = acc[e];
  }
  __syncthreads();
  const float* ccb = cc_old + (size_t)b * Cc * Dd;
  float* chn = ch_new + (size_t)b * Cc * Dd;
  float* ccn = cc_new + (size_t)b * Cc * Dd;
  for (int i = tid; i < Cc * Dd; i += 128) {
    int r = i / Dd, d = i % Dd;
    float ig = gates[r][d]          + bihc[d]          + bhhc[d];
    float fg = gates[r][Dd + d]     + bihc[Dd + d]     + bhhc[Dd + d];
    float gg = gates[r][2 * Dd + d] + bihc[2 * Dd + d] + bhhc[2 * Dd + d];
    float og = gates[r][3 * Dd + d] + bihc[3 * Dd + d] + bhhc[3 * Dd + d];
    float c2 = sigm(fg) * ccb[i] + sigm(ig) * tanhf(gg);
    ccn[i] = c2;
    chn[i] = sigm(og) * tanhf(c2);
  }
}

// ---------------- generic vertex LSTM: gates = A1@B1 + A2@B2 + bias ----------------
// grid = B*N/16 blocks, 128 threads. bias index = bias[bstride*b + col].
// Optionally emits packed-B copy of the new hidden state (for k_adj next step).
__global__ __launch_bounds__(128) void k_lstm(
    const _Float16* __restrict__ A1h, const _Float16* __restrict__ A2h,
    const _Float16* __restrict__ B1p, const _Float16* __restrict__ B2p,
    const float* __restrict__ bias, int bstride,
    const float* __restrict__ vc_old,
    float* __restrict__ vh_new, _Float16* __restrict__ vhh_new,
    _Float16* __restrict__ vhp_new,   // packed-B output or nullptr
    float* __restrict__ vc_new) {
  __shared__ float gates[16][G4];
  int tile = blockIdx.x;
  int b = tile >> 5;
  int tid = threadIdx.x, lane = tid & 31, wave = tid >> 5;
  int n = lane & 15, hf = lane >> 4;
  const _Float16* A1 = A1h + (size_t)tile * 16 * Dd;
  const _Float16* A2 = A2h + (size_t)tile * 16 * Dd;
  for (int ct = wave * 4; ct < wave * 4 + 4; ++ct) {
    v8f acc = {};
#pragma unroll
    for (int kt = 0; kt < Dd / 32; ++kt) {
      v16h a = ld_a_h(A1 + kt * 32, Dd, lane);
      v16h bb = ld_b_p(B1p, Dd / 32, kt, ct, lane);
      acc = WMMA(a, bb, acc);
      a = ld_a_h(A2 + kt * 32, Dd, lane);
      bb = ld_b_p(B2p, Dd / 32, kt, ct, lane);
      acc = WMMA(a, bb, acc);
    }
    float bv = bias[(size_t)bstride * b + ct * 16 + n];
#pragma unroll
    for (int e = 0; e < 8; ++e) gates[hf * 8 + e][ct * 16 + n] = acc[e] + bv;
  }
  __syncthreads();
  size_t base = (size_t)tile * 16 * Dd;
  for (int i = tid; i < 16 * Dd; i += 128) {
    int r = i / Dd, d = i % Dd;
    float ig = gates[r][d], fg = gates[r][Dd + d];
    float gg = gates[r][2 * Dd + d], og = gates[r][3 * Dd + d];
    float c2 = sigm(fg) * vc_old[base + i] + sigm(ig) * tanhf(gg);
    vc_new[base + i] = c2;
    float hh = sigm(og) * tanhf(c2);
    vh_new[base + i] = hh;
    vhh_new[base + i] = (_Float16)hh;
    if (vhp_new) {
      int g = ((tile & 31) << 4) + r;      // row within batch
      int kt = g >> 5, kr = g & 31;
      int lane2 = ((kr >> 4) << 4) + (d & 15);
      int e = kr & 15;
      int nt = d >> 4;
      vhp_new[(size_t)b * Nn * Dd +
              ((((size_t)nt * (Nn / 32) + kt) * 32 + lane2) << 4) + e] = (_Float16)hh;
    }
  }
}

// ---------------- final vote head ----------------
__global__ void k_vote(const float* __restrict__ vh1,
                       const float* __restrict__ W1, const float* __restrict__ b1,
                       const float* __restrict__ W2, const float* __restrict__ b2,
                       float* __restrict__ pred) {
  int b = blockIdx.x, tid = threadIdx.x;  // 64 threads
  __shared__ float red[64];
  float s = 0.f;
  for (int v = tid; v < Nn; v += 64) {
    const float* x = vh1 + ((size_t)b * Nn + v) * Dd;
    float z2 = b2[0];
    for (int j = 0; j < Dd / 4; ++j) {
      float a = b1[j];
      for (int d = 0; d < Dd; ++d) a += x[d] * W1[d * (Dd / 4) + j];
      z2 += sigm(a) * W2[j];
    }
    s += sigm(z2);
  }
  red[tid] = s;
  __syncthreads();
  for (int o = 32; o > 0; o >>= 1) {
    if (tid < o) red[tid] += red[tid + o];
    __syncthreads();
  }
  if (tid == 0) pred[b] = sigm(red[0] / (float)Nn);
}

// ---------------- host orchestration ----------------
extern "C" void kernel_launch(void* const* d_in, const int* in_sizes, int n_in,
                              void* d_out, int out_size, void* d_ws, size_t ws_size,
                              hipStream_t stream) {
  const float* Mvv     = (const float*)d_in[0];
  const int*   ncol    = (const int*)d_in[1];
  const float* vh0_i   = (const float*)d_in[2];
  const float* vh1_i   = (const float*)d_in[3];
  const float* ch_i    = (const float*)d_in[4];
  const float* Wih0    = (const float*)d_in[5];
  const float* Whh0    = (const float*)d_in[6];
  const float* bih0    = (const float*)d_in[7];
  const float* bhh0    = (const float*)d_in[8];
  const float* Wih1    = (const float*)d_in[9];
  const float* Whh1    = (const float*)d_in[10];
  const float* bih1    = (const float*)d_in[11];
  const float* bhh1    = (const float*)d_in[12];
  const float* Wihc    = (const float*)d_in[13];
  const float* Whhc    = (const float*)d_in[14];
  const float* bihc    = (const float*)d_in[15];
  const float* bhhc    = (const float*)d_in[16];
  const float* cW1     = (const float*)d_in[17];
  const float* cb1     = (const float*)d_in[18];
  const float* cW2     = (const float*)d_in[19];
  const float* cb2     = (const float*)d_in[20];
  const float* vW1     = (const float*)d_in[21];
  const float* vb1     = (const float*)d_in[22];
  const float* vW2     = (const float*)d_in[23];
  const float* vb2     = (const float*)d_in[24];
  const float* voteW1  = (const float*)d_in[25];
  const float* voteb1  = (const float*)d_in[26];
  const float* voteW2  = (const float*)d_in[27];
  const float* voteb2  = (const float*)d_in[28];
  (void)in_sizes; (void)n_in; (void)out_size; (void)ws_size;

  char* w = (char*)d_ws;
  size_t off = 0;
  auto alloc = [&](size_t bytes) -> char* {
    char* p = w + off;
    off += (bytes + 255) & ~(size_t)255;
    return p;
  };
  const size_t BND = (size_t)Bb * Nn * Dd;
  const size_t BCD = (size_t)Bb * Cc * Dd;

  _Float16* mvvh   = (_Float16*)alloc((size_t)Bb * Nn * Nn * 2);
  _Float16* mvh    = (_Float16*)alloc(BND * 2);
  float* vh0f[2], *vh1f[2], *vc0f[2], *vc1f[2], *chf[2], *ccf[2];
  _Float16* vh0h[2], *vh1h[2], *vh1pk[2];
  for (int i = 0; i < 2; ++i) {
    vh0f[i]  = (float*)alloc(BND * 4);
    vh1f[i]  = (float*)alloc(BND * 4);
    vc0f[i]  = (float*)alloc(BND * 4);
    vc1f[i]  = (float*)alloc(BND * 4);
    vh0h[i]  = (_Float16*)alloc(BND * 2);
    vh1h[i]  = (_Float16*)alloc(BND * 2);
    vh1pk[i] = (_Float16*)alloc(BND * 2);   // packed-B layout for k_adj
    chf[i]   = (float*)alloc(BCD * 4);
    ccf[i]   = (float*)alloc(BCD * 4);
  }
  float* partial = (float*)alloc((size_t)Bb * (Nn / 16) * Dd * 4);
  float* vsum    = (float*)alloc((size_t)Bb * Dd * 4);
  float* vbias   = (float*)alloc((size_t)Bb * G4 * 4);
  _Float16* vW1p = (_Float16*)alloc((size_t)Dd * HP * 2);
  _Float16* vW2p = (_Float16*)alloc((size_t)HP * Dd * 2);
  _Float16* cW1p = (_Float16*)alloc((size_t)Dd * HP * 2);
  _Float16* cW2p = (_Float16*)alloc((size_t)HP * Dd * 2);
  float* vb1p    = (float*)alloc(HP * 4);
  float* cb1p    = (float*)alloc(HP * 4);
  float* b1sum   = (float*)alloc(G4 * 4);
  _Float16* W0m_p  = (_Float16*)alloc((size_t)Dd * G4 * 2);
  _Float16* Whh0_p = (_Float16*)alloc((size_t)Dd * G4 * 2);
  _Float16* Wih1_p = (_Float16*)alloc((size_t)Dd * G4 * 2);
  _Float16* Whh1_p = (_Float16*)alloc((size_t)Dd * G4 * 2);
  _Float16* Wihc_p = (_Float16*)alloc((size_t)Dd * G4 * 2);
  _Float16* Whhc_p = (_Float16*)alloc((size_t)Dd * G4 * 2);

  const int TPB = 256;
  // ---- prep ----
  k_cvt_f16<<<4096, TPB, 0, stream>>>(mvvh, Mvv, Bb * Nn * Nn);
  k_cvt_f16<<<512, TPB, 0, stream>>>(vh0h[0], vh0_i, (int)BND);
  k_cvt_f16<<<512, TPB, 0, stream>>>(vh1h[0], vh1_i, (int)BND);
  k_pack_act<<<512, TPB, 0, stream>>>(vh1pk[0], vh1_i);
  k_copy_f32<<<512, TPB, 0, stream>>>(vh0f[0], vh0_i, (int)BND);
  k_copy_f32<<<512, TPB, 0, stream>>>(vh1f[0], vh1_i, (int)BND);
  k_copy_f32<<<16, TPB, 0, stream>>>(chf[0], ch_i, (int)BCD);
  k_zero_f32<<<512, TPB, 0, stream>>>(vc0f[0], (int)BND);
  k_zero_f32<<<512, TPB, 0, stream>>>(vc1f[0], (int)BND);
  k_zero_f32<<<16, TPB, 0, stream>>>(ccf[0], (int)BCD);
  // B-operand packing (zero-padded where H=100 -> HP=128)
  k_pack_b<<<64, TPB, 0, stream>>>(vW1p, vW1, Dd, HP, Dd, Hh);
  k_pack_b<<<64, TPB, 0, stream>>>(vW2p, vW2, HP, Dd, Hh, Dd);
  k_pack_b<<<64, TPB, 0, stream>>>(cW1p, cW1, Dd, HP, Dd, Hh);
  k_pack_b<<<64, TPB, 0, stream>>>(cW2p, cW2, HP, Dd, Hh, Dd);
  k_pack_b<<<64, TPB, 0, stream>>>(W0m_p, Wih0 + (size_t)Dd * G4, Dd, G4, Dd, G4);  // Wih0[D:2D,:]
  k_pack_b<<<64, TPB, 0, stream>>>(Whh0_p, Whh0, Dd, G4, Dd, G4);
  k_pack_b<<<64, TPB, 0, stream>>>(Wih1_p, Wih1, Dd, G4, Dd, G4);
  k_pack_b<<<64, TPB, 0, stream>>>(Whh1_p, Whh1, Dd, G4, Dd, G4);
  k_pack_b<<<64, TPB, 0, stream>>>(Wihc_p, Wihc, Dd, G4, Dd, G4);
  k_pack_b<<<64, TPB, 0, stream>>>(Whhc_p, Whhc, Dd, G4, Dd, G4);
  k_pad_f32vec<<<1, HP, 0, stream>>>(vb1p, vb1, Hh, HP);
  k_pad_f32vec<<<1, HP, 0, stream>>>(cb1p, cb1, Hh, HP);
  k_addvec<<<1, G4, 0, stream>>>(b1sum, bih1, bhh1, G4);

  const int MT = Bb * Nn / 16;  // 2048 row tiles
  // ---- T recurrent steps ----
  for (int t = 0; t < Tt; ++t) {
    int cur = t & 1, nxt = cur ^ 1;
    k_adj<<<MT, 128, 0, stream>>>(mvvh, vh1pk[cur], mvh);
    k_vmsg<<<MT, 128, 0, stream>>>(vh1h[cur], vW1p, vb1p, vW2p, vb2, partial);
    k_red<<<Bb, Dd, 0, stream>>>(partial, vsum);
    k_color<<<Bb, 128, 0, stream>>>(chf[cur], ccf[cur], chf[nxt], ccf[nxt], vsum, ncol,
                                    cW1p, cb1p, cW2p, cb2, Wihc_p, Whhc_p, bihc, bhhc,
                                    Wih0, bih0, bhh0, vbias);
    // layer 0: x = [mc|mv] (mc part folded into vbias), h = vh0_old
    k_lstm<<<MT, 128, 0, stream>>>(mvh, vh0h[cur], W0m_p, Whh0_p, vbias, G4,
                                   vc0f[cur], vh0f[nxt], vh0h[nxt], (_Float16*)nullptr,
                                   vc0f[nxt]);
    // layer 1: x = vh0_old (pre-update), h = vh1_old; also emit packed vh1 for next k_adj
    k_lstm<<<MT, 128, 0, stream>>>(vh0h[cur], vh1h[cur], Wih1_p, Whh1_p, b1sum, 0,
                                   vc1f[cur], vh1f[nxt], vh1h[nxt], vh1pk[nxt],
                                   vc1f[nxt]);
  }
  // final vh1 lives in slot Tt&1 == 0
  k_vote<<<Bb, 64, 0, stream>>>(vh1f[Tt & 1], voteW1, voteb1, voteW2, voteb2, (float*)d_out);
}
